// AnalysisTransform_81475529605508
// MI455X (gfx1250) — matrix operand order repaired and checked by reading, not compiled, not run on hardware
//
#include <hip/hip_runtime.h>

// ---------------------------------------------------------------------------
// CDNA5 (gfx1250) implementation of the sparse AnalysisTransform.
// All 5x5x5 convs with C_in in {32,64,96} run as implicit GEMM on
// v_wmma_f32_16x16x32_bf16 (wave32), f32 accumulate, bias/beta folded into
// the accumulator init. Activations live in bf16 channel-last; weights are
// pre-packed into exact per-lane WMMA B-fragment order and staged per-block
// into LDS with double-buffered async global->LDS copies (ASYNCcnt).
// ---------------------------------------------------------------------------

typedef __attribute__((ext_vector_type(16))) __bf16 v16bf;
typedef __attribute__((ext_vector_type(8)))  float  v8f;
typedef __attribute__((ext_vector_type(4)))  unsigned int u32x4;
typedef __attribute__((ext_vector_type(4)))  int int4v;   // builtin's pointee type

union AFrag { v16bf v; unsigned int u[8]; };
union BFrag { v16bf v; u32x4 q[2]; };

#if __has_builtin(__builtin_amdgcn_global_load_async_to_lds_b128) && \
    __has_builtin(__builtin_amdgcn_s_wait_asynccnt)
#define HAVE_ASYNC 1
#define WAIT_ASYNC(n) __builtin_amdgcn_s_wait_asynccnt(n)
#else
#define HAVE_ASYNC 0
#define WAIT_ASYNC(n)
#endif

__device__ __forceinline__ unsigned short f2bf(float f) {
  unsigned int u = __float_as_uint(f);
  unsigned int r = u + 0x7fffu + ((u >> 16) & 1u);   // round-to-nearest-even
  return (unsigned short)(r >> 16);
}

// ---------------------------------------------------------------------------
// q = mean(Q_F, axis=1)  -> d_out q section; scale[b] = q[b][0]
// ---------------------------------------------------------------------------
__global__ void qmean_kernel(const float* __restrict__ QF,
                             float* __restrict__ qout,
                             float* __restrict__ scale) {
  int j = threadIdx.x;
  if (j < 4) {
    int b = j >> 1, c = j & 1;
    float s = 0.f;
    for (int i = 0; i < 1024; ++i) s += QF[(b * 1024 + i) * 2 + c];
    s *= (1.0f / 1024.0f);
    qout[b * 2 + c] = s;
    if (c == 0) scale[b] = s;
  }
}

// ---------------------------------------------------------------------------
// m1 = pool(mask) (64^3 -> 32^3); counts c0 = sum(mask), c1 = sum(m1)
// mask is a numpy bool array -> 1 byte per element on device
// ---------------------------------------------------------------------------
__global__ void pool_mask0_kernel(const unsigned char* __restrict__ mask,
                                  float* __restrict__ m1,
                                  int* __restrict__ c0, int* __restrict__ c1) {
  int idx = blockIdx.x * blockDim.x + threadIdx.x;
  if (idx >= 2 * 32768) return;
  int mv = idx & 32767, b = idx >> 15;
  int ox = mv & 31, oy = (mv >> 5) & 31, oz = mv >> 10;
  int cnt = 0;
  for (int dz = 0; dz < 2; ++dz)
    for (int dy = 0; dy < 2; ++dy)
      for (int dx = 0; dx < 2; ++dx) {
        size_t a = (((size_t)b * 64 + (oz * 2 + dz)) * 64 + (oy * 2 + dy)) * 64
                 + (ox * 2 + dx);
        cnt += mask[a] ? 1 : 0;
      }
  m1[idx] = cnt ? 1.f : 0.f;
  if (cnt) { atomicAdd(c1 + b, 1); atomicAdd(c0 + b, cnt); }
}

// generic stride-2 max pooling of a float mask; optional occupancy counter
__global__ void pool_f_kernel(const float* __restrict__ src,
                              float* __restrict__ dst,
                              int* __restrict__ ctr, int din, int total) {
  int idx = blockIdx.x * blockDim.x + threadIdx.x;
  if (idx >= total) return;
  int dout = din >> 1, ovol = dout * dout * dout;
  int mv = idx % ovol, b = idx / ovol;
  int ox = mv % dout, oy = (mv / dout) % dout, oz = mv / (dout * dout);
  float v = 0.f;
  for (int dz = 0; dz < 2; ++dz)
    for (int dy = 0; dy < 2; ++dy)
      for (int dx = 0; dx < 2; ++dx) {
        float s = src[(((size_t)b * din + (oz * 2 + dz)) * din + (oy * 2 + dy)) * din
                      + (ox * 2 + dx)];
        v = fmaxf(v, s);
      }
  dst[idx] = v;
  if (ctr && v > 0.f) atomicAdd(ctr + b, 1);
}

__global__ void kwrite_kernel(const int* __restrict__ ctr, int* __restrict__ kout) {
  if (threadIdx.x == 0 && blockIdx.x == 0) {
    kout[0] = ctr[4]; kout[1] = ctr[5];   // c2
    kout[2] = ctr[2]; kout[3] = ctr[3];   // c1
    kout[4] = ctr[0]; kout[5] = ctr[1];   // c0
  }
}

// ---------------------------------------------------------------------------
// Weight packer: OIDHW f32 -> bf16 in exact per-lane WMMA B-fragment order.
// Slot (((tap*KC + kc)*CT + ct)*32 + lane)*16 + i holds
//   W[co = ct*16 + (lane&15)][ci = kc*32 + i + (lane>=16)*16][tap]
// so a lane's 16 halfs of a B fragment are 32 contiguous bytes.
// ---------------------------------------------------------------------------
template <int CIN, int COUT>
__global__ void pack_w_kernel(const float* __restrict__ w,
                              unsigned short* __restrict__ out) {
  constexpr int KC = CIN / 32, CT = COUT / 16;
  int idx = blockIdx.x * blockDim.x + threadIdx.x;
  int total = 125 * KC * CT * 512;
  if (idx >= total) return;
  int i = idx & 15;
  int lane = (idx >> 4) & 31;
  int rest = idx >> 9;
  int ct = rest % CT; rest /= CT;
  int kc = rest % KC;
  int t  = rest / KC;
  int ci = kc * 32 + i + ((lane >> 4) << 4);
  int co = ct * 16 + (lane & 15);
  out[idx] = f2bf(w[((size_t)co * CIN + ci) * 125 + t]);
}

// ---------------------------------------------------------------------------
// Layer 1 (C_in=4): plain VALU conv (5% of FLOPs, WMMA-unfriendly K).
// t1 = (conv(x, w1) + b1) * m1 ; also emits bf16 channel-last t1^2 for GDN1.
// ---------------------------------------------------------------------------
__global__ void layer1_kernel(const float* __restrict__ x,
                              const float* __restrict__ w1,
                              const float* __restrict__ b1,
                              const float* __restrict__ m1,
                              float* __restrict__ t1,
                              unsigned short* __restrict__ t1sq) {
  int idx = blockIdx.x * blockDim.x + threadIdx.x;       // b*32*32768 + co*32768 + mv
  if (idx >= 2 * 32 * 32768) return;
  int mv = idx & 32767, co = (idx >> 15) & 31, b = idx >> 20;
  int ox = mv & 31, oy = (mv >> 5) & 31, oz = mv >> 10;
  float s = 0.f;
  for (int t = 0; t < 125; ++t) {
    int tz = t / 25, ty = (t / 5) % 5, tx = t % 5;
    int iz = oz * 2 + tz - 2, iy = oy * 2 + ty - 2, ix = ox * 2 + tx - 2;
    if ((unsigned)iz < 64u && (unsigned)iy < 64u && (unsigned)ix < 64u) {
      size_t base = ((size_t)b * 4) * 262144 + ((size_t)iz * 64 + iy) * 64 + ix;
      const float* wp = w1 + (size_t)co * 4 * 125 + t;
#pragma unroll
      for (int ci = 0; ci < 4; ++ci)
        s += x[base + (size_t)ci * 262144] * wp[(size_t)ci * 125];
    }
  }
  float tv = (s + b1[co]) * m1[b * 32768 + mv];
  t1[idx] = tv;
  t1sq[((size_t)(b * 32768 + mv)) * 32 + co] = f2bf(tv * tv);
}

// ---------------------------------------------------------------------------
// WMMA implicit-GEMM 5x5x5 conv, v2:
//  * block = 4 waves, all sharing one 32-out-channel column pair (ct2);
//    wave w handles voxel tile vt = vtb*4 + w.
//  * each wave holds TWO accumulators (2 x 16 out channels) -> 2 WMMAs per
//    A fragment.
//  * per-tap packed B fragments (KC*2 KB) are staged into LDS once per block,
//    double-buffered with GLOBAL_LOAD_ASYNC_TO_LDS_B128 + s_wait_asynccnt
//    (fallback: VGPR->ds_store staging), then read back as ds_load_b128.
// MODE 0: conv -> t=(acc)*mask, writes f32 NCDHW t and bf16 CL t^2.
// MODE 1: GDN  -> y = t*rsqrt(acc)*scale*mask (acc seeded with beta) -> bf16 CL.
// MODE 2: final conv4 -> (acc)*mask f32 NCDHW into d_out.
// ---------------------------------------------------------------------------
template <int CIN, int COUT, int ID, int OD, int STRIDE, int MODE>
__global__ __launch_bounds__(128)
void conv5_wmma_kernel(const unsigned short* __restrict__ act,   // [B][ID^3][CIN] bf16
                       const unsigned char* __restrict__ wpk,    // packed fragments
                       const float* __restrict__ bias0,          // bias or beta [COUT]
                       const float* __restrict__ tpre,           // f32 t (GDN only)
                       const float* __restrict__ mk,             // [B][OD^3] mask
                       const float* __restrict__ scale,          // [B] (GDN only)
                       float* __restrict__ of32,                 // NCDHW f32 out
                       unsigned short* __restrict__ ocl) {       // bf16 CL out
  constexpr int KC = CIN / 32, CT = COUT / 16, CT2 = CT / 2;
  constexpr int OVOL = OD * OD * OD, IVOL = ID * ID * ID;
  constexpr int NVT = OVOL / 16;
  constexpr int LOG = (OD == 8) ? 3 : ((OD == 16) ? 4 : 5);
  constexpr int STAGE = KC * 2 * 1024;             // bytes of B per tap per block

  __shared__ u32x4 smem_v[2 * STAGE / 16];
  unsigned char* smem = (unsigned char*)smem_v;

  int tid = threadIdx.x;
  int lane = tid & 31;
  int wid = tid >> 5;

  int bk = blockIdx.x;
  int b   = bk / (CT2 * (NVT / 4));
  int r0  = bk % (CT2 * (NVT / 4));
  int ct2 = r0 / (NVT / 4);
  int vtb = r0 % (NVT / 4);
  int vt  = vtb * 4 + wid;

  int mrow = vt * 16 + (lane & 15);                // A-matrix row (out voxel)
  int ox = mrow & (OD - 1), oy = (mrow >> LOG) & (OD - 1), oz = mrow >> (2 * LOG);
  int hi8 = (lane >> 4) * 8;                       // upper half-wave K offset

  float bv0 = bias0[ct2 * 32 + (lane & 15)];
  float bv1 = bias0[ct2 * 32 + 16 + (lane & 15)];
  v8f acc0 = {bv0, bv0, bv0, bv0, bv0, bv0, bv0, bv0};
  v8f acc1 = {bv1, bv1, bv1, bv1, bv1, bv1, bv1, bv1};

  const unsigned short* actb = act + (size_t)b * IVOL * CIN;

  // stage tap t's B fragments (both column halves, all kc) into LDS buffer
  auto stage = [&](int t, int bufsel) {
    unsigned char* dstbase = smem + bufsel * STAGE;
#pragma unroll
    for (int it = 0; it < KC; ++it) {
      int r = (it * 128 + tid) * 16;               // byte offset in stage region
      int blk = r >> 10;                           // which 1KB fragment: kc*2+h
      int within = r & 1023;
      size_t srcoff =
          ((size_t)(t * KC + (blk >> 1)) * CT + (ct2 * 2 + (blk & 1))) * 1024 + within;
#if HAVE_ASYNC
      __builtin_amdgcn_global_load_async_to_lds_b128(
          (int4v*)(wpk + srcoff), (int4v*)(dstbase + r), 0, 0);
#else
      *(u32x4*)(dstbase + r) = *(const u32x4*)(wpk + srcoff);
#endif
      __builtin_prefetch(wpk + srcoff + (size_t)KC * CT * 1024, 0, 2);
    }
  };

  // run one tap's 2*KC WMMAs against the staged buffer
  auto compute_tap = [&](int t, const unsigned char* sb) {
    int tz = t / 25, ty = (t / 5) % 5, tx = t % 5;
    int iz = oz * STRIDE + tz - 2, iy = oy * STRIDE + ty - 2, ix = ox * STRIDE + tx - 2;
    bool inb = (unsigned)iz < (unsigned)ID && (unsigned)iy < (unsigned)ID &&
               (unsigned)ix < (unsigned)ID;
    const unsigned short* ap = actb + ((size_t)((iz * ID + iy) * ID + ix)) * CIN;
#pragma unroll
    for (int kc = 0; kc < KC; ++kc) {
      AFrag a;
      if (inb) {
        const unsigned short* apk = ap + kc * 32 + hi8 * 2;
#pragma unroll
        for (int j = 0; j < 8; ++j) {
          int K2 = 2 * j;                          // K pairs: 0,2,4,...,14 (contiguous)
          a.u[j] = *(const unsigned int*)(apk + K2);
        }
      } else {
#pragma unroll
        for (int j = 0; j < 8; ++j) a.u[j] = 0u;
      }
      const u32x4* l0 = (const u32x4*)(sb + (kc * 2 + 0) * 1024) + lane * 2;
      BFrag w0; w0.q[0] = l0[0]; w0.q[1] = l0[1];
      acc0 = __builtin_amdgcn_wmma_f32_16x16x32_bf16(
          false, a.v, false, w0.v, (short)0, acc0, false, false);
      const u32x4* l1 = (const u32x4*)(sb + (kc * 2 + 1) * 1024) + lane * 2;
      BFrag w1; w1.q[0] = l1[0]; w1.q[1] = l1[1];
      acc1 = __builtin_amdgcn_wmma_f32_16x16x32_bf16(
          false, a.v, false, w1.v, (short)0, acc1, false, false);
    }
  };

  stage(0, 0);
  for (int t = 0; t < 124; ++t) {
    stage(t + 1, (t + 1) & 1);
    WAIT_ASYNC(KC);                                // stage t complete, t+1 in flight
    __syncthreads();
    compute_tap(t, smem + (t & 1) * STAGE);
    __syncthreads();                               // buf reusable for stage t+2
  }
  WAIT_ASYNC(0);
  __syncthreads();
  compute_tap(124, smem + (124 & 1) * STAGE);

  // epilogue: D rows M = r + (lane>=16)*8; columns n = ct2*32 + h*16 + (lane&15)
  float sc = (MODE == 1) ? scale[b] : 0.f;
#pragma unroll
  for (int h = 0; h < 2; ++h) {
    int n = ct2 * 32 + h * 16 + (lane & 15);
#pragma unroll
    for (int r = 0; r < 8; ++r) {
      int mv = vt * 16 + r + hi8;
      float mval = mk[(size_t)b * OVOL + mv];
      size_t oidx = ((size_t)(b * COUT + n)) * OVOL + mv;
      size_t cidx = ((size_t)(b * OVOL + mv)) * COUT + n;
      float av = h ? acc1[r] : acc0[r];
      if constexpr (MODE == 0) {
        float tv = av * mval;
        of32[oidx] = tv;
        ocl[cidx] = f2bf(tv * tv);
      } else if constexpr (MODE == 1) {
        float tv = tpre[oidx];
        float y = tv * rsqrtf(av) * sc * mval;
        ocl[cidx] = f2bf(y);
      } else {
        of32[oidx] = av * mval;
      }
    }
  }
}

// ---------------------------------------------------------------------------
extern "C" void kernel_launch(void* const* d_in, const int* in_sizes, int n_in,
                              void* d_out, int out_size, void* d_ws, size_t ws_size,
                              hipStream_t stream) {
  (void)in_sizes; (void)n_in; (void)out_size; (void)ws_size;
  const float*         x_feat = (const float*)d_in[0];
  const unsigned char* mask   = (const unsigned char*)d_in[1];   // numpy bool
  const float* QF   = (const float*)d_in[2];
  const float* w1   = (const float*)d_in[3];
  const float* b1   = (const float*)d_in[4];
  const float* w2   = (const float*)d_in[5];
  const float* b2   = (const float*)d_in[6];
  const float* w3   = (const float*)d_in[7];
  const float* b3   = (const float*)d_in[8];
  const float* w4   = (const float*)d_in[9];
  const float* b4   = (const float*)d_in[10];
  const float* beta1  = (const float*)d_in[11];
  const float* gamma1 = (const float*)d_in[12];
  const float* beta2  = (const float*)d_in[13];
  const float* gamma2 = (const float*)d_in[14];
  const float* beta3  = (const float*)d_in[15];
  const float* gamma3 = (const float*)d_in[16];

  char* ws = (char*)d_ws;
  size_t off = 0;
  auto carve = [&](size_t bytes) -> void* {
    off = (off + 255) & ~(size_t)255;
    void* p = ws + off;
    off += bytes;
    return p;
  };

  int*   ctr   = (int*)carve(6 * sizeof(int));            // c0[2] c1[2] c2[2]
  float* scale = (float*)carve(2 * sizeof(float));
  float* m1 = (float*)carve((size_t)2 * 32768 * 4);
  float* m2 = (float*)carve((size_t)2 * 4096 * 4);
  float* m3 = (float*)carve((size_t)2 * 512 * 4);
  float* t1 = (float*)carve((size_t)2 * 32 * 32768 * 4);
  unsigned short* t1sq = (unsigned short*)carve((size_t)2 * 32768 * 32 * 2);
  unsigned short* y1   = (unsigned short*)carve((size_t)2 * 32768 * 32 * 2);
  float* t2 = (float*)carve((size_t)2 * 64 * 4096 * 4);
  unsigned short* t2sq = (unsigned short*)carve((size_t)2 * 4096 * 64 * 2);
  unsigned short* y2   = (unsigned short*)carve((size_t)2 * 4096 * 64 * 2);
  float* t3 = (float*)carve((size_t)2 * 96 * 512 * 4);
  unsigned short* t3sq = (unsigned short*)carve((size_t)2 * 512 * 96 * 2);
  unsigned short* y3   = (unsigned short*)carve((size_t)2 * 512 * 96 * 2);
  unsigned short* wp_g1 = (unsigned short*)carve((size_t)125 * 1 * 2 * 1024);
  unsigned short* wp_w2 = (unsigned short*)carve((size_t)125 * 1 * 4 * 1024);
  unsigned short* wp_g2 = (unsigned short*)carve((size_t)125 * 2 * 4 * 1024);
  unsigned short* wp_w3 = (unsigned short*)carve((size_t)125 * 2 * 6 * 1024);
  unsigned short* wp_g3 = (unsigned short*)carve((size_t)125 * 3 * 6 * 1024);
  unsigned short* wp_w4 = (unsigned short*)carve((size_t)125 * 3 * 8 * 1024);

  float* out_x = (float*)d_out;                    // [2][128][8][8][8]
  float* out_q = out_x + 131072;                   // [1][2][2]
  int*   out_k = (int*)(out_x + 131076);           // [3][2] int32

  (void)hipMemsetAsync(ctr, 0, 6 * sizeof(int), stream);

  qmean_kernel<<<1, 64, 0, stream>>>(QF, out_q, scale);

  pool_mask0_kernel<<<2 * 32768 / 256, 256, 0, stream>>>(mask, m1, ctr, ctr + 2);
  pool_f_kernel<<<(2 * 4096 + 255) / 256, 256, 0, stream>>>(m1, m2, ctr + 4, 32, 2 * 4096);
  pool_f_kernel<<<(2 * 512 + 255) / 256, 256, 0, stream>>>(m2, m3, nullptr, 16, 2 * 512);
  kwrite_kernel<<<1, 1, 0, stream>>>(ctr, out_k);

  pack_w_kernel<32, 32><<<(125 * 1 * 2 * 512 + 255) / 256, 256, 0, stream>>>(gamma1, wp_g1);
  pack_w_kernel<32, 64><<<(125 * 1 * 4 * 512 + 255) / 256, 256, 0, stream>>>(w2, wp_w2);
  pack_w_kernel<64, 64><<<(125 * 2 * 4 * 512 + 255) / 256, 256, 0, stream>>>(gamma2, wp_g2);
  pack_w_kernel<64, 96><<<(125 * 2 * 6 * 512 + 255) / 256, 256, 0, stream>>>(w3, wp_w3);
  pack_w_kernel<96, 96><<<(125 * 3 * 6 * 512 + 255) / 256, 256, 0, stream>>>(gamma3, wp_g3);
  pack_w_kernel<96, 128><<<(125 * 3 * 8 * 512 + 255) / 256, 256, 0, stream>>>(w4, wp_w4);

  layer1_kernel<<<2 * 32 * 32768 / 256, 256, 0, stream>>>(x_feat, w1, b1, m1, t1, t1sq);

  // GDN1: den = beta1 + conv(gamma1, t1^2); y1 = t1*rsqrt(den)*q0*m1
  conv5_wmma_kernel<32, 32, 32, 32, 1, 1><<<2 * 1 * 512, 128, 0, stream>>>(
      t1sq, (const unsigned char*)wp_g1, beta1, t1, m1, scale, nullptr, y1);
  // conv2: t2 = (conv(y1, w2) + b2) * m2
  conv5_wmma_kernel<32, 64, 32, 16, 2, 0><<<2 * 2 * 64, 128, 0, stream>>>(
      y1, (const unsigned char*)wp_w2, b2, nullptr, m2, nullptr, t2, t2sq);
  // GDN2
  conv5_wmma_kernel<64, 64, 16, 16, 1, 1><<<2 * 2 * 64, 128, 0, stream>>>(
      t2sq, (const unsigned char*)wp_g2, beta2, t2, m2, scale, nullptr, y2);
  // conv3: t3 = (conv(y2, w3) + b3) * m3
  conv5_wmma_kernel<64, 96, 16, 8, 2, 0><<<2 * 3 * 8, 128, 0, stream>>>(
      y2, (const unsigned char*)wp_w3, b3, nullptr, m3, nullptr, t3, t3sq);
  // GDN3
  conv5_wmma_kernel<96, 96, 8, 8, 1, 1><<<2 * 3 * 8, 128, 0, stream>>>(
      t3sq, (const unsigned char*)wp_g3, beta3, t3, m3, scale, nullptr, y3);
  // conv4 (stride 1): x = (conv(y3, w4) + b4) * m3  -> final output
  conv5_wmma_kernel<96, 128, 8, 8, 1, 2><<<2 * 4 * 8, 128, 0, stream>>>(
      y3, (const unsigned char*)wp_w4, b4, nullptr, m3, nullptr, out_x, nullptr);
}
